// DualBranchHGAT_11158325035086
// MI455X (gfx1250) — matrix operand order repaired and checked by reading, not compiled, not run on hardware
//
#include <hip/hip_runtime.h>

typedef __attribute__((ext_vector_type(16))) _Float16 v16h;
typedef __attribute__((ext_vector_type(8)))  _Float16 v8h;
typedef __attribute__((ext_vector_type(8)))  float    v8f;

namespace {
constexpr int kNSku = 100000;
constexpr int kNWh  = 5000;
constexpr int kNSup = 2000;
constexpr int kDIn  = 128;
constexpr int kDOut = 64;
constexpr int kH    = 4;
constexpr int kDh   = 16;

// Workspace layout (offsets in floats). Buffers reused sequentially per edge type.
constexpr long O_H_SRC   = 0;                                   // 100000*64
constexpr long O_H_DST   = O_H_SRC  + (long)kNSku * kDOut;      // 5000*64 (cross dst)
constexpr long O_S_SRC   = O_H_DST  + (long)kNWh  * kDOut;      // 100000*4
constexpr long O_S_DST   = O_S_SRC  + (long)kNSku * kH;         // 100000*4
constexpr long O_DMAX    = O_S_DST  + (long)kNSku * kH;         // 100000*4 (ordered uint)
constexpr long O_DEN     = O_DMAX   + (long)kNSku * kH;         // 100000*4
constexpr long O_WBUF    = O_DEN    + (long)kNSku * kH;         // 400000*4 (exp weights)
constexpr long O_OUT_SKU = O_WBUF   + 400000L * kH;             // 100000*64
constexpr long O_OUT_WH  = O_OUT_SKU + (long)kNSku * kDOut;     // 5000*64
constexpr long O_OUT_SUP = O_OUT_WH  + (long)kNWh  * kDOut;     // 2000*64
// f16 copies of node features (sizes below are in FLOAT units = halves/2)
constexpr long O_X16_SKU = O_OUT_SUP + (long)kNSup * kDOut;     // 100000*128/2
constexpr long O_X16_WH  = O_X16_SKU + (long)kNSku * kDIn / 2;  // 5000*128/2
constexpr long O_X16_SUP = O_X16_WH  + (long)kNWh  * kDIn / 2;  // 2000*128/2
constexpr long WS_END    = O_X16_SUP + (long)kNSup * kDIn / 2;  // ~95 MB total
} // namespace

__device__ __forceinline__ v8f wmma_f16(v16h a, v16h b, v8f c) {
  // D = A(16x32 f16) x B(32x16 f16) + C(16x16 f32)
  return __builtin_amdgcn_wmma_f32_16x16x32_f16(false, a, false, b, (short)0, c,
                                                false, false);
}

// A-fragment: two contiguous 8-half (16B) runs at p and p+16 (ISA 16-bit A layout).
__device__ __forceinline__ v16h load_a_frag(const _Float16* __restrict__ p) {
  const v8h lo = *(const v8h*)p;
  const v8h hi = *(const v8h*)(p + 16);
  return __builtin_shufflevector(lo, hi, 0, 1, 2, 3, 4, 5, 6, 7,
                                 8, 9, 10, 11, 12, 13, 14, 15);
}

// Ordered-uint encoding of float for atomicMax (monotone; 0 encodes below -inf).
__device__ __forceinline__ unsigned f2ord(float f) {
  int b = __float_as_int(f);
  return (b >= 0) ? ((unsigned)b | 0x80000000u) : ~(unsigned)b;
}
__device__ __forceinline__ float ord2f(unsigned u) {
  int b = (u & 0x80000000u) ? (int)(u & 0x7fffffffu) : (int)~u;
  return __int_as_float(b);
}

// ---------------------------------------------------------------------------
// One-time f32 -> f16 conversion of node features (4 elements per thread).
// ---------------------------------------------------------------------------
__global__ void k_cvt_f16(const float* __restrict__ src,
                          _Float16* __restrict__ dst, long n) {
  const long i = ((long)blockIdx.x * blockDim.x + threadIdx.x) * 4;
  if (i >= n) return;
  const float4 v = *(const float4*)(src + i);
  v8h* o2 = nullptr; (void)o2;
  _Float16 h0 = (_Float16)v.x, h1 = (_Float16)v.y;
  _Float16 h2 = (_Float16)v.z, h3 = (_Float16)v.w;
  typedef __attribute__((ext_vector_type(4))) _Float16 v4h;
  v4h o = {h0, h1, h2, h3};
  *(v4h*)(dst + i) = o;
}

// ---------------------------------------------------------------------------
// h = A16[N,128] @ W[128,64], f16 WMMA, f32 accumulate.
// 256 threads = 8 waves; one wave -> 16 rows -> 128 rows/block.
// W staged in LDS pre-swizzled into B-fragment layout (2x ds_load_b128/frag).
// ---------------------------------------------------------------------------
__global__ void __launch_bounds__(256)
k_proj_gemm(const _Float16* __restrict__ A16, const float* __restrict__ W,
            float* __restrict__ Hout, int N) {
  __shared__ v16h Wf[16 * 32]; // 16 fragments x 32 lanes x 32B = 16 KB
  {
    _Float16* Wl = (_Float16*)Wf;
    for (int j = threadIdx.x; j < 16 * 32 * 16; j += 256) {
      const int f = j >> 9;         // fragment (512 halves each)
      const int l = (j >> 4) & 31;  // lane
      const int i = j & 15;         // element within lane
      const int q = f >> 2, c = f & 3;
      const int k = q * 32 + (l >> 4) * 8 + (i & 7) + (i >> 3) * 16;
      const int n = c * 16 + (l & 15);
      Wl[j] = (_Float16)W[k * kDOut + n];
    }
  }
  __syncthreads();

  const int wave = threadIdx.x >> 5;
  const int lane = threadIdx.x & 31;
  const int m    = lane & 15;      // A row within tile / C column
  const int grp  = lane >> 4;
  const int row0 = blockIdx.x * 128 + wave * 16;

  int rowA = row0 + m;
  if (rowA >= N) rowA = N - 1;     // clamp; out-of-range rows never stored

  v16h afr[4];
#pragma unroll
  for (int q = 0; q < 4; ++q)
    afr[q] = load_a_frag(A16 + (long)rowA * kDIn + q * 32 + grp * 8);

  const bool full = (row0 + 16 <= N); // wave-uniform fast path
#pragma unroll
  for (int c = 0; c < 4; ++c) {
    v8f acc = {};
#pragma unroll
    for (int q = 0; q < 4; ++q)
      acc = wmma_f16(afr[q], Wf[(q * 4 + c) * 32 + lane], acc);
    const int n = c * 16 + m;
    float* op = Hout + (long)(row0 + 8 * grp) * kDOut + n;
    if (full) {
#pragma unroll
      for (int r = 0; r < 8; ++r) op[(long)r * kDOut] = acc[r];
    } else {
#pragma unroll
      for (int r = 0; r < 8; ++r) {
        const int rr = row0 + r + 8 * grp;
        if (rr < N) Hout[(long)rr * kDOut + n] = acc[r];
      }
    }
  }
}

// ---------------------------------------------------------------------------
// Per-node head scalars: S[n,h] = dot(h[n, h*16 : h*16+16], a[0:16])
// ---------------------------------------------------------------------------
__global__ void k_head_dot(const float* __restrict__ Hm,
                           const float* __restrict__ av,
                           float* __restrict__ S, int N) {
  int idx = blockIdx.x * blockDim.x + threadIdx.x;
  if (idx >= N * kH) return;
  const int node = idx >> 2;
  const int h    = idx & 3;
  const float* hp = Hm + (long)node * kDOut + h * kDh;
  float s = 0.f;
#pragma unroll
  for (int i = 0; i < kDh; ++i) s += hp[i] * av[i];
  S[idx] = s;
}

// ---------------------------------------------------------------------------
// Edge pass 1: per-destination segment max of elu(score)
// ---------------------------------------------------------------------------
__global__ void k_edge_max(const int* __restrict__ ei, int E,
                           const float* __restrict__ Ssrc,
                           const float* __restrict__ Sdst,
                           unsigned* __restrict__ dmax) {
  int idx = blockIdx.x * blockDim.x + threadIdx.x;
  if (idx >= E * kH) return;
  const int e = idx >> 2, h = idx & 3;
  const int s = ei[e], d = ei[E + e];
  const float x  = Ssrc[(long)s * kH + h] + Sdst[(long)d * kH + h];
  const float el = (x > 0.f) ? x : (__expf(x) - 1.f);
  atomicMax(dmax + (long)d * kH + h, f2ord(el));
}

// ---------------------------------------------------------------------------
// Edge pass 2: w = exp(e - max), accumulate denominator, stash w
// ---------------------------------------------------------------------------
__global__ void k_edge_expsum(const int* __restrict__ ei, int E,
                              const float* __restrict__ Ssrc,
                              const float* __restrict__ Sdst,
                              const unsigned* __restrict__ dmax,
                              float* __restrict__ den,
                              float* __restrict__ wbuf) {
  int idx = blockIdx.x * blockDim.x + threadIdx.x;
  if (idx >= E * kH) return;
  const int e = idx >> 2, h = idx & 3;
  const int s = ei[e], d = ei[E + e];
  const float x  = Ssrc[(long)s * kH + h] + Sdst[(long)d * kH + h];
  const float el = (x > 0.f) ? x : (__expf(x) - 1.f);
  const float w  = __expf(el - ord2f(dmax[(long)d * kH + h]));
  wbuf[idx] = w;
  atomicAdd(den + (long)d * kH + h, w);
}

// ---------------------------------------------------------------------------
// Edge pass 3: out[dst, head] += alpha * h_src[src, head]
// ---------------------------------------------------------------------------
__global__ void k_edge_scatter(const int* __restrict__ ei, int E,
                               const float* __restrict__ wbuf,
                               const float* __restrict__ den,
                               const float* __restrict__ Hsrc,
                               float* __restrict__ outAcc) {
  int idx = blockIdx.x * blockDim.x + threadIdx.x;
  if (idx >= E * kH) return;
  const int e = idx >> 2, h = idx & 3;
  const int s = ei[e], d = ei[E + e];
  const float* hs = Hsrc + (long)s * kDOut + h * kDh;
  __builtin_prefetch(hs, 0, 1); // global_prefetch_b8
  const float alpha = wbuf[idx] / (den[(long)d * kH + h] + 1e-12f);
  float* od = outAcc + (long)d * kDOut + h * kDh;
#pragma unroll
  for (int j = 0; j < kDh; ++j) atomicAdd(od + j, alpha * hs[j]);
}

// ---------------------------------------------------------------------------
// finish: y = [oa|ob]@proj_W + proj_b + x@res_W, then LayerNorm. K=192 WMMA.
// 256 threads = 8 waves = 128 rows/block; pre-swizzled B fragments in LDS.
// ---------------------------------------------------------------------------
__global__ void __launch_bounds__(256)
k_finish(const _Float16* __restrict__ X16, // [N,128] f16 copy
         const float* __restrict__ O,      // [N,64] active branch accumulator
         const float* __restrict__ Wp,     // 64x64 half of proj_W for this branch
         const float* __restrict__ pb, const float* __restrict__ Wr, // res_W 128x64
         const float* __restrict__ g, const float* __restrict__ bb,
         float* __restrict__ Y, int N) {
  __shared__ v16h Wf[24 * 32]; // 24 fragments (q=0..5, c=0..3) = 24 KB
  {
    _Float16* Wl = (_Float16*)Wf;
    for (int j = threadIdx.x; j < 24 * 32 * 16; j += 256) {
      const int f = j >> 9;
      const int l = (j >> 4) & 31;
      const int i = j & 15;
      const int q = f >> 2, c = f & 3;
      const int k = q * 32 + (l >> 4) * 8 + (i & 7) + (i >> 3) * 16; // 0..191 combined K
      const int n = c * 16 + (l & 15);
      const float v = (k < kDOut) ? Wp[k * kDOut + n]
                                  : Wr[(k - kDOut) * kDOut + n];
      Wl[j] = (_Float16)v;
    }
  }
  __syncthreads();

  const int wave = threadIdx.x >> 5;
  const int lane = threadIdx.x & 31;
  const int m    = lane & 15;
  const int grp  = lane >> 4;
  const int row0 = blockIdx.x * 128 + wave * 16;
  int rowA = row0 + m;
  if (rowA >= N) rowA = N - 1;

  v8f accs[4] = {};
#pragma unroll
  for (int q = 0; q < 6; ++q) {   // K chunks: 0-1 from O (K=64), 2-5 from X16 (K=128)
    v16h a;
    if (q < 2) {
      const float* p = O + (long)rowA * kDOut + q * 32 + grp * 8;
#pragma unroll
      for (int i = 0; i < 8; ++i) {
        a[i]     = (_Float16)p[i];
        a[8 + i] = (_Float16)p[16 + i];
      }
    } else {
      a = load_a_frag(X16 + (long)rowA * kDIn + (q - 2) * 32 + grp * 8);
    }
#pragma unroll
    for (int c = 0; c < 4; ++c)
      accs[c] = wmma_f16(a, Wf[(q * 4 + c) * 32 + lane], accs[c]);
  }

  float pbv[4], gv[4], bv[4];
#pragma unroll
  for (int c = 0; c < 4; ++c) {
    const int n = c * 16 + m;
    pbv[c] = pb[n]; gv[c] = g[n]; bv[c] = bb[n];
  }

  const bool full = (row0 + 16 <= N); // wave-uniform fast path
#pragma unroll
  for (int r = 0; r < 8; ++r) {
    float v[4];
    float sum = 0.f;
#pragma unroll
    for (int c = 0; c < 4; ++c) { v[c] = accs[c][r] + pbv[c]; sum += v[c]; }
#pragma unroll
    for (int off = 1; off < 16; off <<= 1) sum += __shfl_xor(sum, off, 32);
    const float mu = sum * (1.0f / 64.0f);
    float sq = 0.f;
#pragma unroll
    for (int c = 0; c < 4; ++c) { v[c] -= mu; sq += v[c] * v[c]; }
#pragma unroll
    for (int off = 1; off < 16; off <<= 1) sq += __shfl_xor(sq, off, 32);
    const float rstd = rsqrtf(sq * (1.0f / 64.0f) + 1e-5f);
    const int rr = row0 + r + 8 * grp;
    if (full || rr < N) {
#pragma unroll
      for (int c = 0; c < 4; ++c)
        Y[(long)rr * kDOut + c * 16 + m] = v[c] * rstd * gv[c] + bv[c];
    }
  }
}

// ---------------------------------------------------------------------------
extern "C" void kernel_launch(void* const* d_in, const int* in_sizes, int n_in,
                              void* d_out, int out_size, void* d_ws, size_t ws_size,
                              hipStream_t stream) {
  (void)n_in; (void)out_size; (void)ws_size;
  const float* x_sku = (const float*)d_in[0];
  const float* x_wh  = (const float*)d_in[1];
  const float* x_sup = (const float*)d_in[2];
  const float* Wt[6]; const float* at[6];
  for (int t = 0; t < 6; ++t) {
    Wt[t] = (const float*)d_in[3 + 2 * t];
    at[t] = (const float*)d_in[4 + 2 * t];
  }
  const float* proj_W = (const float*)d_in[15];
  const float* proj_b = (const float*)d_in[16];
  const float* ln_g   = (const float*)d_in[17];
  const float* ln_b   = (const float*)d_in[18];
  const float* res_W  = (const float*)d_in[19];
  const int* ei[6]; int E[6];
  for (int t = 0; t < 6; ++t) {
    ei[t] = (const int*)d_in[20 + t];
    E[t]  = in_sizes[20 + t] / 2;
  }

  float* ws = (float*)d_ws;
  float* y  = (float*)d_out;
  _Float16* x16_sku = (_Float16*)(ws + O_X16_SKU);
  _Float16* x16_wh  = (_Float16*)(ws + O_X16_WH);
  _Float16* x16_sup = (_Float16*)(ws + O_X16_SUP);

  // One-time f16 conversions of node features.
  {
    const long n1 = (long)kNSku * kDIn, n2 = (long)kNWh * kDIn,
               n3 = (long)kNSup * kDIn;
    k_cvt_f16<<<(int)((n1 / 4 + 255) / 256), 256, 0, stream>>>(x_sku, x16_sku, n1);
    k_cvt_f16<<<(int)((n2 / 4 + 255) / 256), 256, 0, stream>>>(x_wh, x16_wh, n2);
    k_cvt_f16<<<(int)((n3 / 4 + 255) / 256), 256, 0, stream>>>(x_sup, x16_sup, n3);
  }

  // Zero persistent branch accumulators.
  hipMemsetAsync(ws + O_OUT_SKU, 0,
                 (size_t)(O_X16_SKU - O_OUT_SKU) * sizeof(float), stream);

  auto run_type = [&](int t, const _Float16* xsrc, int Nsrc, const _Float16* xdst,
                      int Ndst, float* outAcc) {
    float* h_src = ws + O_H_SRC;
    float* h_dst = h_src;
    k_proj_gemm<<<(Nsrc + 127) / 128, 256, 0, stream>>>(xsrc, Wt[t], h_src, Nsrc);
    if (xdst != xsrc) {
      h_dst = ws + O_H_DST;
      k_proj_gemm<<<(Ndst + 127) / 128, 256, 0, stream>>>(xdst, Wt[t], h_dst, Ndst);
    }
    float* s_src = ws + O_S_SRC;
    float* s_dst = ws + O_S_DST;
    k_head_dot<<<(Nsrc * kH + 255) / 256, 256, 0, stream>>>(h_src, at[t], s_src, Nsrc);
    k_head_dot<<<(Ndst * kH + 255) / 256, 256, 0, stream>>>(h_dst, at[t] + kDh,
                                                            s_dst, Ndst);
    unsigned* dmax = (unsigned*)(ws + O_DMAX);
    float* den  = ws + O_DEN;
    float* wbuf = ws + O_WBUF;
    hipMemsetAsync(dmax, 0, (size_t)Ndst * kH * sizeof(unsigned), stream);
    hipMemsetAsync(den,  0, (size_t)Ndst * kH * sizeof(float), stream);
    const int nthr = E[t] * kH;
    k_edge_max<<<(nthr + 255) / 256, 256, 0, stream>>>(ei[t], E[t], s_src, s_dst,
                                                       dmax);
    k_edge_expsum<<<(nthr + 255) / 256, 256, 0, stream>>>(ei[t], E[t], s_src, s_dst,
                                                          dmax, den, wbuf);
    k_edge_scatter<<<(nthr + 255) / 256, 256, 0, stream>>>(ei[t], E[t], wbuf, den,
                                                           h_src, outAcc);
  };

  // Branch A: 4 intra sku->sku types summed into the sku accumulator.
  run_type(0, x16_sku, kNSku, x16_sku, kNSku, ws + O_OUT_SKU);
  run_type(1, x16_sku, kNSku, x16_sku, kNSku, ws + O_OUT_SKU);
  run_type(2, x16_sku, kNSku, x16_sku, kNSku, ws + O_OUT_SKU);
  run_type(3, x16_sku, kNSku, x16_sku, kNSku, ws + O_OUT_SKU);
  // Branch B: cross types.
  run_type(4, x16_sku, kNSku, x16_wh,  kNWh,  ws + O_OUT_WH);
  run_type(5, x16_sku, kNSku, x16_sup, kNSup, ws + O_OUT_SUP);

  // finish(): sku uses proj_W rows 0..63 (ob==0); wh/sup use rows 64..127 (oa==0).
  k_finish<<<(kNSku + 127) / 128, 256, 0, stream>>>(
      x16_sku, ws + O_OUT_SKU, proj_W, proj_b, res_W, ln_g, ln_b, y, kNSku);
  k_finish<<<(kNWh + 127) / 128, 256, 0, stream>>>(
      x16_wh, ws + O_OUT_WH, proj_W + 64 * 64, proj_b, res_W, ln_g, ln_b,
      y + (long)kNSku * kDOut, kNWh);
  k_finish<<<(kNSup + 127) / 128, 256, 0, stream>>>(
      x16_sup, ws + O_OUT_SUP, proj_W + 64 * 64, proj_b, res_W, ln_g, ln_b,
      y + (long)(kNSku + kNWh) * kDOut, kNSup);
}
